// ATTN_17944373363076
// MI455X (gfx1250) — matrix operand order repaired and checked by reading, not compiled, not run on hardware
//
#include <hip/hip_runtime.h>
#include <hip/hip_bf16.h>
#include <math.h>

typedef __attribute__((ext_vector_type(16))) _Float16 v16h;
typedef __attribute__((ext_vector_type(8)))  float    v8f;

#define D_NODE 128
#define DQ 256
#define DK 384
#define KN 16
#define DH 128

// ---------------------------------------------------------------------------
// Fragment helpers (CDNA5 wave32 WMMA; A/B use contiguous k order (valid as A and B match), C/D per CDNA5 ISA §7.12.2)
// ---------------------------------------------------------------------------

// A-matrix 16x32 f16: lane<16 -> row=lane, K = s*32 + 0..15
//                     lane>=16 -> row=lane-16, K = s*32 + 16..31
__device__ inline v16h ldsA(const _Float16* base, int ldk, int ks) {
    int lane = threadIdx.x & 31;
    const _Float16* p = base + (size_t)(lane & 15) * ldk + ks * 32 + ((lane & 16) ? 16 : 0);
    v16h f;
#pragma unroll
    for (int i = 0; i < 16; ++i) f[i] = p[i];
    return f;
}

// B-matrix 32x16 f16 from row-major weight W[OUT][IN] (B = W^T, so B column n = W row n):
// lane<16 -> col=lane, K = s*32 + 0..15 ; lane>=16 -> col=lane-16, K = s*32 + 16..31
__device__ inline v16h ldgB(const _Float16* W, int ldin, int colbase, int ks) {
    int lane = threadIdx.x & 31;
    const _Float16* p = W + (size_t)(colbase + (lane & 15)) * ldin + ks * 32 + ((lane & 16) ? 16 : 0);
    v16h f;
#pragma unroll
    for (int i = 0; i < 16; ++i) f[i] = p[i];
    return f;
}

// D = A(16xK) * W^T tile (Kx16) accumulated over KSTEPS k-steps of 32
template <int KSTEPS>
__device__ inline v8f wmma_rowtile(const _Float16* A, int lda,
                                   const _Float16* W, int ldin, int colbase) {
    v8f acc = {};
#pragma unroll
    for (int s = 0; s < KSTEPS; ++s) {
        v16h a = ldsA(A, lda, s);
        v16h b = ldgB(W, ldin, colbase, s);
        acc = __builtin_amdgcn_wmma_f32_16x16x32_f16(false, a, false, b,
                                                     (short)0, acc, false, false);
    }
    return acc;
}

// C/D 16x16 f32: lane<16 -> col=lane, rows 0..7 in v[0..7]; lane>=16 -> col=lane-16, rows 8..15
__device__ inline void stD_f16(_Float16* base, int ldd, int colbase, v8f acc,
                               const float* bias, float scale, bool relu) {
    int lane = threadIdx.x & 31;
    int col = colbase + (lane & 15);
    int m0 = (lane & 16) ? 8 : 0;
    float b = bias ? bias[col] : 0.f;
#pragma unroll
    for (int r = 0; r < 8; ++r) {
        float v = (acc[r] + b) * scale;
        if (relu) v = v > 0.f ? v : 0.f;
        base[(size_t)(m0 + r) * ldd + col] = (_Float16)v;
    }
}

__device__ inline void stD_out(float* out, int n0, int colbase, v8f acc,
                               const float* bias, int N) {
    int lane = threadIdx.x & 31;
    int col = colbase + (lane & 15);
    int m0 = (lane & 16) ? 8 : 0;
    float b = bias[col];
#pragma unroll
    for (int r = 0; r < 8; ++r) {
        int n = n0 + m0 + r;
        if (n < N) out[(size_t)n * 128 + col] = acc[r] + b;
    }
}

// ---------------------------------------------------------------------------
// Kernel 0: convert all weight matrices fp32 -> f16 into workspace
// ---------------------------------------------------------------------------
__global__ void tgat_prep(const float* Wq, const float* Wk, const float* Wv,
                          const float* Wo, const float* W1, const float* W2,
                          _Float16* wh) {
    int i = blockIdx.x * blockDim.x + threadIdx.x;
    const int n0 = 65536;            // Wq 256x256
    const int n1 = n0 + 98304;       // Wk 256x384
    const int n2 = n1 + 98304;       // Wv 256x384
    const int n3 = n2 + 65536;       // Wo 256x256
    const int n4 = n3 + 49152;       // W1 128x384
    const int n5 = n4 + 16384;       // W2 128x128
    if (i < n0)      wh[i] = (_Float16)Wq[i];
    else if (i < n1) wh[i] = (_Float16)Wk[i - n0];
    else if (i < n2) wh[i] = (_Float16)Wv[i - n1];
    else if (i < n3) wh[i] = (_Float16)Wo[i - n2];
    else if (i < n4) wh[i] = (_Float16)W1[i - n3];
    else if (i < n5) wh[i] = (_Float16)W2[i - n4];
}

// ---------------------------------------------------------------------------
// Kernel 1: fused TGAT attention. 1 block = 16 nodes, 256 threads = 8 waves.
// ---------------------------------------------------------------------------
__global__ __launch_bounds__(256)
void tgat_main(const float* __restrict__ node_emb, const float* __restrict__ dst_emb,
               const float* __restrict__ edge_feat, const float* __restrict__ timestamps,
               const float* __restrict__ last_update, const int* __restrict__ src_idx,
               const float* __restrict__ w_t, const float* __restrict__ b_t,
               const float* __restrict__ bq, const float* __restrict__ bk,
               const float* __restrict__ bv, const float* __restrict__ bo,
               const float* __restrict__ b1, const float* __restrict__ b2,
               const _Float16* __restrict__ wq16, const _Float16* __restrict__ wk16,
               const _Float16* __restrict__ wv16, const _Float16* __restrict__ wo16,
               const _Float16* __restrict__ w1_16, const _Float16* __restrict__ w2_16,
               float* __restrict__ out_h, float* __restrict__ att_unnorm, int N) {
    __shared__ _Float16 sA[16 * DK];    // A staging (query / per-node C / merge hidden)
    __shared__ _Float16 sKV[16 * 512];  // current node: kproj(0:256) | vproj(256:512)
    __shared__ float    sCTX[16 * DQ];  // per-node attention context
    __shared__ _Float16 sQX[16 * DK];   // q (ld 256) during loop; X=[dst|h_before] (ld 384) after
    __shared__ float    sScore[32];
    __shared__ float    sP[32];

    const int tid = threadIdx.x;
    const int wave = tid >> 5;
    const int n0 = blockIdx.x * 16;
    const float qscale = 0.08838834764831845f;  // 1/sqrt(dh=128)

    // ---------------- Phase Q: q = (concat(dst, cos(b_t)) @ Wq^T + bq) * scale ---------
    for (int idx = tid; idx < 16 * DQ; idx += 256) {
        int i = idx >> 8, c = idx & 255;
        int n = n0 + i;
        float v = (c < 128) ? ((n < N) ? dst_emb[(size_t)n * 128 + c] : 0.f)
                            : cosf(b_t[c - 128]);
        sA[idx] = (_Float16)v;
    }
    __syncthreads();
#pragma unroll
    for (int t = 0; t < 2; ++t) {
        int ct = wave * 2 + t;
        v8f acc = wmma_rowtile<8>(sA, DQ, wq16, DQ, ct * 16);
        stD_f16(sQX, DQ, ct * 16, acc, bq, qscale, false);
    }
    __syncthreads();

    // ---------------- Node loop: per-node edge projections + attention -----------------
    for (int i = 0; i < 16; ++i) {
        int n = n0 + i;
        // Build C = [node_emb[src] | edge_feat | cos(dt*w+b)]  (16 edges x 384) as f16
        for (int idx = tid; idx < KN * DK; idx += 256) {
            int e = idx / DK, c = idx - e * DK;
            float v = 0.f;
            if (n < N) {
                int src = src_idx[(size_t)n * KN + e];
                if (c < 128)       v = node_emb[(size_t)src * 128 + c];
                else if (c < 256)  v = edge_feat[((size_t)n * KN + e) * 128 + (c - 128)];
                else {
                    float dt = timestamps[(size_t)n * KN + e] - last_update[src];
                    int j = c - 256;
                    v = cosf(fmaf(dt, w_t[j], b_t[j]));
                }
            }
            sA[idx] = (_Float16)v;
        }
        __syncthreads();
        // kproj / vproj: 32 output col-tiles over 8 waves, 12 k-steps each
#pragma unroll
        for (int t = 0; t < 4; ++t) {
            int ct = wave * 4 + t;
            bool isV = ct >= 16;
            const _Float16* W = isV ? wv16 : wk16;
            const float* bias = isV ? bv : bk;
            int cb = (isV ? (ct - 16) : ct) * 16;
            v8f acc = wmma_rowtile<12>(sA, DK, W, DK, cb);
            stD_f16(sKV, 512, (isV ? 256 : 0) + cb, acc, bias, 1.0f, false);
        }
        __syncthreads();
        // scores[h][k] = qh . kh
        if (tid < 32) {
            int h = tid >> 4, k = tid & 15;
            const _Float16* qp = sQX + i * DQ + h * DH;
            const _Float16* kp = sKV + k * 512 + h * DH;
            float s = 0.f;
            for (int d = 0; d < DH; ++d) s += (float)qp[d] * (float)kp[d];
            sScore[tid] = s;
        }
        __syncthreads();
        // softmax over K per head
        if (tid < 2) {
            int h = tid;
            float m = -1e30f;
            for (int k = 0; k < 16; ++k) m = fmaxf(m, sScore[h * 16 + k]);
            float e[16], sum = 0.f;
            for (int k = 0; k < 16; ++k) { e[k] = expf(sScore[h * 16 + k] - m); sum += e[k]; }
            float inv = 1.f / sum;
            for (int k = 0; k < 16; ++k) sP[h * 16 + k] = e[k] * inv;
        }
        __syncthreads();
        // ctx[d] = sum_k p[h(d)][k] * v[k][d] ; att = mean over heads (unnormalized)
        {
            int d = tid;
            if (d < DQ) {
                int h = d >> 7;
                float acc = 0.f;
                for (int k = 0; k < 16; ++k)
                    acc += sP[h * 16 + k] * (float)sKV[k * 512 + 256 + d];
                sCTX[i * DQ + d] = acc;
            }
            if (tid < 16 && n < N)
                att_unnorm[(size_t)n * 16 + tid] = 0.5f * (sP[tid] + sP[16 + tid]);
        }
        __syncthreads();
    }

    // ---------------- Phase O: h_before = ctx @ Wo^T + bo -> X[:,128:384] ----------------
    for (int idx = tid; idx < 16 * DQ; idx += 256) sA[idx] = (_Float16)sCTX[idx];
    __syncthreads();
#pragma unroll
    for (int t = 0; t < 2; ++t) {
        int ct = wave * 2 + t;
        v8f acc = wmma_rowtile<8>(sA, DQ, wo16, DQ, ct * 16);
        stD_f16(sQX, DK, 128 + ct * 16, acc, bo, 1.0f, false);
    }
    // X[:,0:128] = dst_emb (disjoint from the D-stores above)
    for (int idx = tid; idx < 16 * 128; idx += 256) {
        int i2 = idx >> 7, c = idx & 127;
        int n = n0 + i2;
        sQX[(size_t)i2 * DK + c] = (_Float16)((n < N) ? dst_emb[(size_t)n * 128 + c] : 0.f);
    }
    __syncthreads();

    // ---------------- Merge layer 1: relu(X @ W1^T + b1) -> sA (16x128 f16) -------------
    {
        v8f acc = wmma_rowtile<12>(sQX, DK, w1_16, DK, wave * 16);
        stD_f16(sA, 128, wave * 16, acc, b1, 1.0f, true);
    }
    __syncthreads();
    // ---------------- Merge layer 2: hid @ W2^T + b2 -> out_h ---------------------------
    {
        v8f acc = wmma_rowtile<4>(sA, 128, w2_16, 128, wave * 16);
        stD_out(out_h, n0, wave * 16, acc, b2, N);
    }
}

// ---------------------------------------------------------------------------
// Kernels 2-4: attn_map = scatter(att / sum(att)) via eid (deterministic)
// ---------------------------------------------------------------------------
__global__ void tgat_fill(float* out_att, int total) {
    int i = blockIdx.x * blockDim.x + threadIdx.x;
    if (i < total) out_att[i] = -1.0f;
}

__global__ void tgat_sum(const float* __restrict__ att, int total, float* ws_sum) {
    __shared__ float red[1024];
    float s = 0.f;
    for (int i = threadIdx.x; i < total; i += 1024) s += att[i];
    red[threadIdx.x] = s;
    __syncthreads();
    for (int off = 512; off > 0; off >>= 1) {
        if (threadIdx.x < off) red[threadIdx.x] += red[threadIdx.x + off];
        __syncthreads();
    }
    if (threadIdx.x == 0) *ws_sum = red[0];
}

__global__ void tgat_scale(const float* __restrict__ att, const int* __restrict__ eid,
                           float* __restrict__ out_att, const float* __restrict__ ws_sum,
                           int total) {
    int i = blockIdx.x * blockDim.x + threadIdx.x;
    if (i < total) {
        int e = eid[i];
        if (e >= 0 && e < total) out_att[e] = att[i] * (1.f / *ws_sum);
    }
}

// ---------------------------------------------------------------------------
extern "C" void kernel_launch(void* const* d_in, const int* in_sizes, int n_in,
                              void* d_out, int out_size, void* d_ws, size_t ws_size,
                              hipStream_t stream) {
    (void)n_in; (void)out_size; (void)ws_size;
    const float* node_emb    = (const float*)d_in[0];
    const float* dst_emb     = (const float*)d_in[1];
    const float* edge_feat   = (const float*)d_in[2];
    const float* timestamps  = (const float*)d_in[3];
    const float* last_update = (const float*)d_in[4];
    const int*   src_idx     = (const int*)d_in[5];
    const int*   eid         = (const int*)d_in[6];
    const float* w_t = (const float*)d_in[7];
    const float* b_t = (const float*)d_in[8];
    const float* Wq = (const float*)d_in[9];  const float* bq = (const float*)d_in[10];
    const float* Wk = (const float*)d_in[11]; const float* bk = (const float*)d_in[12];
    const float* Wv = (const float*)d_in[13]; const float* bv = (const float*)d_in[14];
    const float* Wo = (const float*)d_in[15]; const float* bo = (const float*)d_in[16];
    const float* W1 = (const float*)d_in[17]; const float* b1 = (const float*)d_in[18];
    const float* W2 = (const float*)d_in[19]; const float* b2 = (const float*)d_in[20];

    const int N = in_sizes[1] / 128;   // dst_emb rows
    const int total = N * KN;

    // workspace layout (16B-aligned):
    // [0,16)          float sum accumulator
    // [16, 786448)    f16 weights (Wq|Wk|Wv|Wo|W1|W2)
    // [786448, ...)   unnormalized attention, N*K floats
    char* ws = (char*)d_ws;
    float*    ws_sum = (float*)ws;
    _Float16* wh     = (_Float16*)(ws + 16);
    _Float16* wq16 = wh;
    _Float16* wk16 = wq16 + 65536;
    _Float16* wv16 = wk16 + 98304;
    _Float16* wo16 = wv16 + 98304;
    _Float16* w1_16 = wo16 + 65536;
    _Float16* w2_16 = w1_16 + 49152;
    float* att_ws = (float*)(ws + 786448);

    float* out_h   = (float*)d_out;
    float* out_att = out_h + (size_t)N * 128;

    tgat_prep<<<1536, 256, 0, stream>>>(Wq, Wk, Wv, Wo, W1, W2, wh);

    int nblk = (N + 15) / 16;
    tgat_main<<<nblk, 256, 0, stream>>>(node_emb, dst_emb, edge_feat, timestamps,
                                        last_update, src_idx, w_t, b_t,
                                        bq, bk, bv, bo, b1, b2,
                                        wq16, wk16, wv16, wo16, w1_16, w2_16,
                                        out_h, att_ws, N);

    tgat_fill<<<(total + 255) / 256, 256, 0, stream>>>(out_att, total);
    tgat_sum<<<1, 1024, 0, stream>>>(att_ws, total, ws_sum);
    tgat_scale<<<(total + 255) / 256, 256, 0, stream>>>(att_ws, eid, out_att, ws_sum, total);
}
